// MultiHeadAttention_2473901162691
// MI455X (gfx1250) — matrix operand order repaired and checked by reading, not compiled, not run on hardware
//
#include <hip/hip_runtime.h>
#include <hip/hip_bf16.h>

// MHA for MI455X (gfx1250): f16 WMMA everywhere, f32 accumulate.
// B=4, S=2048, D=1024, H=16, DK=64.

typedef __attribute__((ext_vector_type(16))) _Float16 v16h;
typedef __attribute__((ext_vector_type(8)))  float    v8f;

constexpr int B  = 4;
constexpr int S  = 2048;
constexpr int D  = 1024;
constexpr int H  = 16;
constexpr int DK = 64;

// Padded LDS row strides (bank-conflict avoidance on the 64x4B banks)
constexpr int SCP = S + 4;   // f32 score rows: +16B pad
constexpr int PFP = S + 8;   // f16 prob rows:  +16B pad -> lane m hits banks 4m..4m+3

// ---------------- WMMA helpers ----------------

__device__ __forceinline__ v8f wmma32(v16h a, v16h b, v8f c) {
  // D = A(16x32 f16) * B(32x16 f16) + C(16x16 f32)
  return __builtin_amdgcn_wmma_f32_16x16x32_f16(
      /*neg_a=*/false, a, /*neg_b=*/false, b,
      /*c_mod=*/(short)0, c, /*reuse_a=*/false, /*reuse_b=*/false);
}

// A-fragment, f16 source (global or LDS), row-major; p -> (row m, col kblk).
// Lanes 0-15: kbase=0, lanes 16-31: kbase=8; element j<8 -> K=kbase+j,
// j>=8 -> K=kbase+16+(j-8)  (ISA 7.12.2 16-bit A layout).
__device__ __forceinline__ v16h ldA_f16(const _Float16* p, int hi) {
  const _Float16* q = p + hi * 8;
  union { v16h h; float4 f[2]; } u;
  u.f[0] = *reinterpret_cast<const float4*>(q);        // K = kbase..kbase+7
  u.f[1] = *reinterpret_cast<const float4*>(q + 16);   // K = kbase+16..kbase+23
  return u.h;
}

// B-fragment, f16 source where B[K][N] = src[n][k] (row n contiguous along K).
// Same K layout as ldA_f16: kbase=8*hi, j<8 -> kbase+j, j>=8 -> kbase+16+(j-8).
__device__ __forceinline__ v16h ldB_f16(const _Float16* p, int hi) {
  const _Float16* q = p + hi * 8;
  union { v16h h; float4 f[2]; } u;
  u.f[0] = *reinterpret_cast<const float4*>(q);
  u.f[1] = *reinterpret_cast<const float4*>(q + 16);
  return u.h;
}

// A-fragment from f32 source, converted to f16 in-register.
__device__ __forceinline__ v16h ldA_f32(const float* __restrict__ p, int hi) {
  const float* q = p + hi * 8;
  float4 x0 = *reinterpret_cast<const float4*>(q);
  float4 x1 = *reinterpret_cast<const float4*>(q + 4);
  float4 x2 = *reinterpret_cast<const float4*>(q + 16);
  float4 x3 = *reinterpret_cast<const float4*>(q + 20);
  v16h a;
  a[0]=(_Float16)x0.x;  a[1]=(_Float16)x0.y;  a[2]=(_Float16)x0.z;  a[3]=(_Float16)x0.w;
  a[4]=(_Float16)x1.x;  a[5]=(_Float16)x1.y;  a[6]=(_Float16)x1.z;  a[7]=(_Float16)x1.w;
  a[8]=(_Float16)x2.x;  a[9]=(_Float16)x2.y;  a[10]=(_Float16)x2.z; a[11]=(_Float16)x2.w;
  a[12]=(_Float16)x3.x; a[13]=(_Float16)x3.y; a[14]=(_Float16)x3.z; a[15]=(_Float16)x3.w;
  return a;
}

// B-fragment from f32 source (row contiguous along K), converted to f16.
__device__ __forceinline__ v16h ldB_f32(const float* __restrict__ p, int hi) {
  const float* q = p + hi * 8;
  float4 x0 = *reinterpret_cast<const float4*>(q);
  float4 x1 = *reinterpret_cast<const float4*>(q + 4);
  float4 x2 = *reinterpret_cast<const float4*>(q + 16);
  float4 x3 = *reinterpret_cast<const float4*>(q + 20);
  v16h b;
  b[0]=(_Float16)x0.x;  b[1]=(_Float16)x0.y;  b[2]=(_Float16)x0.z;  b[3]=(_Float16)x0.w;
  b[4]=(_Float16)x1.x;  b[5]=(_Float16)x1.y;  b[6]=(_Float16)x1.z;  b[7]=(_Float16)x1.w;
  b[8]=(_Float16)x2.x;  b[9]=(_Float16)x2.y;  b[10]=(_Float16)x2.z; b[11]=(_Float16)x2.w;
  b[12]=(_Float16)x3.x; b[13]=(_Float16)x3.y; b[14]=(_Float16)x3.z; b[15]=(_Float16)x3.w;
  return b;
}

// ---------------- Kernel 1: QKV projections ----------------
// grid = (B*S/16, H, 3), block = 32 (one wave).
// z=0 -> Q (pre-scaled 1/sqrt(DK)), z=1 -> K, z=2 -> V stored transposed [b,h,dk,t].
__global__ void mha_proj_kernel(
    const float* __restrict__ xq, const float* __restrict__ xk, const float* __restrict__ xv,
    const float* __restrict__ Wq, const float* __restrict__ bq,
    const float* __restrict__ Wk, const float* __restrict__ bk,
    const float* __restrict__ Wv, const float* __restrict__ bv,
    _Float16* __restrict__ Qh, _Float16* __restrict__ Kh, _Float16* __restrict__ Vt) {
  const int lane = threadIdx.x;
  const int nlo  = lane & 15;
  const int hi   = (lane < 16) ? 0 : 1;
  const int h    = blockIdx.y;
  const int bs   = blockIdx.x;
  const int b    = bs / (S / 16);
  const int s0   = (bs % (S / 16)) * 16;
  const int which = blockIdx.z;

  const float* x; const float* W; const float* bias;
  if (which == 0)      { x = xq; W = Wq; bias = bq; }
  else if (which == 1) { x = xk; W = Wk; bias = bk; }
  else                 { x = xv; W = Wv; bias = bv; }
  const float scale = (which == 0) ? 0.125f : 1.0f;  // 1/sqrt(64) folded into Q

  v8f acc[4];
#pragma unroll
  for (int nt = 0; nt < 4; ++nt) {
    float bb = bias[h * DK + nt * 16 + nlo];
    v8f c;
#pragma unroll
    for (int r = 0; r < 8; ++r) c[r] = bb;
    acc[nt] = c;
  }

  const float* arow = x + ((size_t)b * S + s0 + nlo) * D;
  for (int kb = 0; kb < D; kb += 32) {
    v16h a = ldA_f32(arow + kb, hi);
#pragma unroll
    for (int nt = 0; nt < 4; ++nt) {
      const float* brow = W + ((size_t)h * DK + nt * 16 + nlo) * D + kb;
      v16h bf = ldB_f32(brow, hi);
      acc[nt] = wmma32(a, bf, acc[nt]);
    }
  }

#pragma unroll
  for (int nt = 0; nt < 4; ++nt) {
#pragma unroll
    for (int r = 0; r < 8; ++r) {
      const int M   = r + hi * 8;
      const int col = nt * 16 + nlo;
      const float val = acc[nt][r] * scale;
      if (which == 2) {
        Vt[(((size_t)b * H + h) * DK + col) * S + s0 + M] = (_Float16)val;
      } else {
        _Float16* dst = (which == 0) ? Qh : Kh;
        dst[(((size_t)b * H + h) * S + s0 + M) * DK + col] = (_Float16)val;
      }
    }
  }
}

// ---------------- Kernel 2: attention + softmax + head-average ----------------
// grid = (B*S/16), block = 256 (8 waves). One block owns a 16-row query tile for
// ALL heads. LDS: padded f32 score block (128.2 KB) + padded f16 prob block
// (64.3 KB) + reduce scratch (~5 KB) ~= 198 KB of the WGP's 320 KB.
// The per-block 16xS head-average rows are owned exclusively -> plain global RMW.
__global__ void mha_attn_kernel(
    const _Float16* __restrict__ Qh, const _Float16* __restrict__ Kh,
    const _Float16* __restrict__ Vt, _Float16* __restrict__ Ch,
    float* __restrict__ avg_out) {
  extern __shared__ float lds[];
  float*     sc    = lds;                               // 16 * SCP f32 (scores->exp)
  _Float16*  pf    = (_Float16*)(sc + 16 * SCP);        // 16 * PFP f16 (probs)
  float*     red   = (float*)(pf + 16 * PFP);           // 16*16 row-reduce scratch
  float*     rstat = red + 256;                         // 16 rowmax + 16 inv-rowsum
  float*     obuf  = rstat + 32;                        // 4 * 256 f32 (O cross-wave)

  const int tid  = threadIdx.x;
  const int w    = tid >> 5;
  const int lane = tid & 31;
  const int nlo  = lane & 15;
  const int hi   = (lane < 16) ? 0 : 1;
  const int bs = blockIdx.x;
  const int b  = bs / (S / 16);
  const int s0 = (bs % (S / 16)) * 16;

  for (int h = 0; h < H; ++h) {
    // ---- prefetch next head's K and V^T (128B line per prefetch, no counters) ----
    if (h + 1 < H) {
      const _Float16* nk = Kh + (((size_t)b * H + (h + 1)) * S) * DK;
      const _Float16* nv = Vt + (((size_t)b * H + (h + 1)) * DK) * S;
#pragma unroll
      for (int j = 0; j < 8; ++j) {
        __builtin_prefetch(nk + ((size_t)tid * 8 + j) * 64, 0, 0);
        __builtin_prefetch(nv + ((size_t)tid * 8 + j) * 64, 0, 0);
      }
    }

    // ---- scores: S_tile = Q_tile (16xDK) x K^T, DK=64 -> two x32 WMMAs/tile ----
    const _Float16* qrow = Qh + (((size_t)b * H + h) * S + s0 + nlo) * DK;
    v16h aq0 = ldA_f16(qrow, hi);
    v16h aq1 = ldA_f16(qrow + 32, hi);
#pragma unroll 1
    for (int i = 0; i < 16; ++i) {          // 128 t-tiles spread over 8 waves
      const int t0 = (w + 8 * i) * 16;
      const _Float16* krow = Kh + (((size_t)b * H + h) * S + t0 + nlo) * DK;
      v16h b0 = ldB_f16(krow, hi);
      v16h b1 = ldB_f16(krow + 32, hi);
      v8f c = {};
      c = wmma32(aq0, b0, c);
      c = wmma32(aq1, b1, c);
#pragma unroll
      for (int r = 0; r < 8; ++r) sc[(r + hi * 8) * SCP + t0 + nlo] = c[r];
    }
    __syncthreads();

    // ---- exact softmax, 16-thread row teams; probs -> f16 LDS + global avg ----
    {
      const int row = tid >> 4, sub = tid & 15;
      float* srow = sc + row * SCP;
      float m = -1e30f;
      for (int c = sub; c < S; c += 16) m = fmaxf(m, srow[c]);
      red[row * 16 + sub] = m;
      __syncthreads();
      if (sub == 0) {
        float mm = red[row * 16];
#pragma unroll
        for (int j = 1; j < 16; ++j) mm = fmaxf(mm, red[row * 16 + j]);
        rstat[row] = mm;
      }
      __syncthreads();
      const float mm = rstat[row];
      float ssum = 0.0f;
      for (int c = sub; c < S; c += 16) {
        float e = __expf(srow[c] - mm);
        srow[c] = e;
        ssum += e;
      }
      red[row * 16 + sub] = ssum;
      __syncthreads();
      if (sub == 0) {
        float t = 0.0f;
#pragma unroll
        for (int j = 0; j < 16; ++j) t += red[row * 16 + j];
        rstat[16 + row] = 1.0f / t;
      }
      __syncthreads();
      const float inv = rstat[16 + row];
      _Float16* prow = pf + row * PFP;
      float* arow = avg_out + ((size_t)b * S + s0 + row) * S;
      for (int c = sub; c < S; c += 16) {
        float p = srow[c] * inv;
        prow[c] = (_Float16)p;
        float contrib = p * (1.0f / H);
        if (h == 0) arow[c] = contrib;      // first head initializes (poison-safe)
        else        arow[c] += contrib;     // exclusive rows: plain RMW, no atomics
      }
    }
    __syncthreads();

    // ---- O = P @ V: 8 waves = 4 dk-tiles x 2 t-halves; reduce pairs via LDS ----
    {
      const int dkt = w & 3, th = w >> 2;
      v8f o = {};
      const _Float16* vrow =
          Vt + (((size_t)b * H + h) * DK + dkt * 16 + nlo) * S + th * (S / 2);
      const _Float16* prow = pf + nlo * PFP + th * (S / 2);
#pragma unroll 1
      for (int tk = 0; tk < S / 2; tk += 32) {
        v16h a  = ldA_f16(prow + tk, hi);   // 2x ds_load_b128, conflict-free
        v16h bv = ldB_f16(vrow + tk, hi);   // V^T rows contiguous along t
        o = wmma32(a, bv, o);
      }
      if (w >= 4) {
#pragma unroll
        for (int r = 0; r < 8; ++r) obuf[(w - 4) * 256 + r * 32 + lane] = o[r];
      }
      __syncthreads();
      if (w < 4) {
#pragma unroll
        for (int r = 0; r < 8; ++r) {
          o[r] += obuf[w * 256 + r * 32 + lane];
          const int M = r + hi * 8;
          Ch[((size_t)b * S + s0 + M) * D + h * DK + w * 16 + nlo] = (_Float16)o[r];
        }
      }
    }
    __syncthreads();   // sc/pf are reused by the next head
  }
}

// ---------------- Kernel 3: output projection out = concat @ Wo^T + bo ----------------
// grid = (B*S/16, D/16), block = 32 (one wave per 16x16 output tile).
__global__ void mha_out_kernel(
    const _Float16* __restrict__ Ch, const float* __restrict__ Wo,
    const float* __restrict__ bo, float* __restrict__ out) {
  const int lane = threadIdx.x;
  const int nlo  = lane & 15;
  const int hi   = (lane < 16) ? 0 : 1;
  const int bs = blockIdx.x;
  const int b  = bs / (S / 16);
  const int s0 = (bs % (S / 16)) * 16;
  const int n0 = blockIdx.y * 16;

  v8f c;
  const float bb = bo[n0 + nlo];
#pragma unroll
  for (int r = 0; r < 8; ++r) c[r] = bb;

  const _Float16* arow = Ch + ((size_t)b * S + s0 + nlo) * D;
  const float*    brow = Wo + (size_t)(n0 + nlo) * D;   // Wo[n][d]: contiguous along K
  for (int kb = 0; kb < D; kb += 32) {
    v16h a  = ldA_f16(arow + kb, hi);
    v16h bf = ldB_f32(brow + kb, hi);
    c = wmma32(a, bf, c);
  }
#pragma unroll
  for (int r = 0; r < 8; ++r)
    out[((size_t)b * S + s0 + r + hi * 8) * D + n0 + nlo] = c[r];
}

// ---------------- launch ----------------
extern "C" void kernel_launch(void* const* d_in, const int* in_sizes, int n_in,
                              void* d_out, int out_size, void* d_ws, size_t ws_size,
                              hipStream_t stream) {
  const float* q  = (const float*)d_in[0];
  const float* k  = (const float*)d_in[1];
  const float* v  = (const float*)d_in[2];
  const float* Wq = (const float*)d_in[3];
  const float* bq = (const float*)d_in[4];
  const float* Wk = (const float*)d_in[5];
  const float* bk = (const float*)d_in[6];
  const float* Wv = (const float*)d_in[7];
  const float* bv = (const float*)d_in[8];
  const float* Wo = (const float*)d_in[9];
  const float* bo = (const float*)d_in[10];

  float* out = (float*)d_out;                      // (B,S,D)
  float* avg = out + (size_t)B * S * D;            // (B,S,S)

  // Workspace: 4 x 16 MB of f16 (Q, K, V^T, concat-heads) = 64 MB.
  _Float16* Qh = (_Float16*)d_ws;
  _Float16* Kh = Qh + (size_t)B * H * S * DK;
  _Float16* Vt = Kh + (size_t)B * H * S * DK;
  _Float16* Ch = Vt + (size_t)B * H * S * DK;

  dim3 g1(B * S / 16, H, 3);
  mha_proj_kernel<<<g1, 32, 0, stream>>>(q, k, v, Wq, bq, Wk, bk, Wv, bv, Qh, Kh, Vt);

  const size_t shmem = (size_t)16 * SCP * sizeof(float)       // f32 scores
                     + (size_t)16 * PFP * sizeof(_Float16)    // f16 probs
                     + (size_t)(256 + 32) * sizeof(float)     // reduce scratch
                     + (size_t)4 * 256 * sizeof(float);       // O cross-wave buf
  mha_attn_kernel<<<dim3(B * S / 16), 256, shmem, stream>>>(Qh, Kh, Vt, Ch, avg);

  mha_out_kernel<<<dim3(B * S / 16, D / 16), 32, 0, stream>>>(Ch, Wo, bo, out);
}
